// QuantizedSSM_12369505812873
// MI455X (gfx1250) — matrix operand order repaired and checked by reading, not compiled, not run on hardware
//
#include <hip/hip_runtime.h>

// ---------------------------------------------------------------------------
// QuantizedSSM for MI455X (gfx1250), wave32 + v_wmma_f32_16x16x32_bf16.
//   x_next = x * A_diag + s_B * (u @ B_bar^T)
//   y      = s_C * (x_next @ C_bar^T) + s_D * (u @ D_bar^T)
// Weights are 4-bit ints stored f32 -> exact in bf16. Activations use a
// hi/lo bf16 split (2 WMMAs per k-step) -> ~fp32 accuracy, memory bound.
// The split is precomputed ONCE per element into a packed u32 stream
// (hi | lo<<16); consumers deswizzle with v_perm_b32 (cheap) instead of
// re-converting in every wave's k-loop.
// ---------------------------------------------------------------------------

#define D_STATE 1024
#define D_IN    256
#define D_OUT   256
#define BATCH   32768

typedef __attribute__((ext_vector_type(16))) __bf16        v16bf;
typedef __attribute__((ext_vector_type(8)))  float         v8f;
typedef __attribute__((ext_vector_type(8)))  unsigned int  v8u;
typedef __attribute__((ext_vector_type(4)))  unsigned int  v4u;

#define HI_SEL 0x05040100u   // v_perm selector: {h0,h1} from two packed words
#define LO_SEL 0x07060302u   // v_perm selector: {l0,l1}

__device__ __forceinline__ unsigned short f32_to_bf16_rne(float f) {
    unsigned int u = __float_as_uint(f);
    u += 0x7FFFu + ((u >> 16) & 1u);
    return (unsigned short)(u >> 16);
}
__device__ __forceinline__ float bf16_to_f32(unsigned short h) {
    return __uint_as_float(((unsigned int)h) << 16);
}
// Packed split of one f32: bf16(hi) | bf16(residual)<<16
__device__ __forceinline__ unsigned int split_pack(float v) {
    unsigned short h = f32_to_bf16_rne(v);
    unsigned short l = f32_to_bf16_rne(v - bf16_to_f32(h));
    return (unsigned int)h | ((unsigned int)l << 16);
}

// ---- Fallback path: build A hi/lo fragments straight from f32 (round-1) ----
__device__ __forceinline__ void cvt_split8(const float* __restrict__ p,
                                           unsigned int* hi_u, unsigned int* lo_u) {
    float4 a = *(const float4*)(p);
    float4 b = *(const float4*)(p + 4);
    float v[8] = {a.x, a.y, a.z, a.w, b.x, b.y, b.z, b.w};
#pragma unroll
    for (int i = 0; i < 4; ++i) {
        unsigned short h0 = f32_to_bf16_rne(v[2 * i]);
        unsigned short h1 = f32_to_bf16_rne(v[2 * i + 1]);
        unsigned short l0 = f32_to_bf16_rne(v[2 * i]     - bf16_to_f32(h0));
        unsigned short l1 = f32_to_bf16_rne(v[2 * i + 1] - bf16_to_f32(h1));
        hi_u[i] = (unsigned int)h0 | ((unsigned int)h1 << 16);
        lo_u[i] = (unsigned int)l0 | ((unsigned int)l1 << 16);
    }
}
__device__ __forceinline__ void load_a_convert(const float* __restrict__ row,
                                               int k0, int half,
                                               v16bf& a_hi, v16bf& a_lo) {
    unsigned int hi[8], lo[8];
    cvt_split8(row + k0 + half * 8,      hi,     lo);
    cvt_split8(row + k0 + 16 + half * 8, hi + 4, lo + 4);
    v8u h, l;
#pragma unroll
    for (int i = 0; i < 8; ++i) { h[i] = hi[i]; l[i] = lo[i]; }
    a_hi = __builtin_bit_cast(v16bf, h);
    a_lo = __builtin_bit_cast(v16bf, l);
}

// ---- Fast path: A fragments from packed split stream via v_perm_b32 -------
__device__ __forceinline__ void load_a_from_split(const unsigned int* __restrict__ s,
                                                  int k0, int half,
                                                  v16bf& a_hi, v16bf& a_lo) {
    v4u w0 = *(const v4u*)(s + k0 + half * 8);
    v4u w1 = *(const v4u*)(s + k0 + half * 8 + 4);
    v4u w2 = *(const v4u*)(s + k0 + 16 + half * 8);
    v4u w3 = *(const v4u*)(s + k0 + 16 + half * 8 + 4);
    v8u h, l;
    h[0] = __builtin_amdgcn_perm(w0[1], w0[0], HI_SEL);
    h[1] = __builtin_amdgcn_perm(w0[3], w0[2], HI_SEL);
    h[2] = __builtin_amdgcn_perm(w1[1], w1[0], HI_SEL);
    h[3] = __builtin_amdgcn_perm(w1[3], w1[2], HI_SEL);
    h[4] = __builtin_amdgcn_perm(w2[1], w2[0], HI_SEL);
    h[5] = __builtin_amdgcn_perm(w2[3], w2[2], HI_SEL);
    h[6] = __builtin_amdgcn_perm(w3[1], w3[0], HI_SEL);
    h[7] = __builtin_amdgcn_perm(w3[3], w3[2], HI_SEL);
    l[0] = __builtin_amdgcn_perm(w0[1], w0[0], LO_SEL);
    l[1] = __builtin_amdgcn_perm(w0[3], w0[2], LO_SEL);
    l[2] = __builtin_amdgcn_perm(w1[1], w1[0], LO_SEL);
    l[3] = __builtin_amdgcn_perm(w1[3], w1[2], LO_SEL);
    l[4] = __builtin_amdgcn_perm(w2[1], w2[0], LO_SEL);
    l[5] = __builtin_amdgcn_perm(w2[3], w2[2], LO_SEL);
    l[6] = __builtin_amdgcn_perm(w3[1], w3[0], LO_SEL);
    l[7] = __builtin_amdgcn_perm(w3[3], w3[2], LO_SEL);
    a_hi = __builtin_bit_cast(v16bf, h);
    a_lo = __builtin_bit_cast(v16bf, l);
}

// Load bf16 weights w[0..7] and w[16..23] (16B-aligned) as a B-matrix fragment.
__device__ __forceinline__ v16bf load_w16(const unsigned short* __restrict__ w) {
    v4u a = *(const v4u*)(w);
    v4u b = *(const v4u*)(w + 16);
    v8u c;
#pragma unroll
    for (int i = 0; i < 4; ++i) { c[i] = a[i]; c[4 + i] = b[i]; }
    return __builtin_bit_cast(v16bf, c);
}

// ---------------------------------------------------------------------------
// Prep A: bf16 weights (exact) + A_diag = -softplus(A_bar * s_A)
// ---------------------------------------------------------------------------
__global__ void __launch_bounds__(256) ssm_prep_kernel(
    const float* __restrict__ A_bar, const float* __restrict__ B_bar,
    const float* __restrict__ C_bar, const float* __restrict__ D_bar,
    const float* __restrict__ sA_ptr,
    unsigned short* __restrict__ Bw, unsigned short* __restrict__ Cw,
    unsigned short* __restrict__ Dw, float* __restrict__ A_diag)
{
    int i = blockIdx.x * blockDim.x + threadIdx.x;
    if (i < D_STATE) {
        float z  = A_bar[i] * (*sA_ptr);
        float sp = fmaxf(z, 0.0f) + log1pf(expf(-fabsf(z)));   // stable softplus
        A_diag[i] = -sp;
    }
    if (i < D_STATE * D_IN) {   // 262144: covers B and C
        Bw[i] = f32_to_bf16_rne(B_bar[i]);
        Cw[i] = f32_to_bf16_rne(C_bar[i]);
    }
    if (i < D_OUT * D_IN) {     // 65536
        Dw[i] = f32_to_bf16_rne(D_bar[i]);
    }
}

// ---------------------------------------------------------------------------
// Prep B: u -> packed hi/lo split stream (4 elems/thread, b128 in/out)
// ---------------------------------------------------------------------------
__global__ void __launch_bounds__(256) ssm_usplit_kernel(
    const float* __restrict__ u, unsigned int* __restrict__ usplit)
{
    size_t i4 = (size_t)blockIdx.x * blockDim.x + threadIdx.x;   // 2,097,152
    float4 v = *(const float4*)(u + i4 * 4);
    uint4 s;
    s.x = split_pack(v.x); s.y = split_pack(v.y);
    s.z = split_pack(v.z); s.w = split_pack(v.w);
    *(uint4*)(usplit + i4 * 4) = s;
}

// ---------------------------------------------------------------------------
// Kernel 1: x_next = x * A_diag + s_B * (u @ B^T). Wave -> 16x64 tile.
// Optionally also emits the packed split of x_next for kernel 2.
// ---------------------------------------------------------------------------
template <bool USE_USPLIT, bool WRITE_XSPLIT>
__global__ void __launch_bounds__(256) ssm_xnext_kernel(
    const float* __restrict__ x, const float* __restrict__ u,
    const unsigned int* __restrict__ usplit,
    const unsigned short* __restrict__ Bw, const float* __restrict__ A_diag,
    const float* __restrict__ sB_ptr,
    float* __restrict__ x_next, unsigned int* __restrict__ xsplit)
{
    const int lane = threadIdx.x & 31;
    const int wave = blockIdx.x * (blockDim.x >> 5) + (threadIdx.x >> 5);
    const int m0   = (wave >> 4) << 4;   // 2048 M-tiles
    const int n0   = (wave & 15) << 6;   // 16 N-groups of 64
    const int half = lane >> 4;
    const int l16  = lane & 15;

    v8f acc[4];
#pragma unroll
    for (int t = 0; t < 4; ++t)
#pragma unroll
        for (int r = 0; r < 8; ++r) acc[t][r] = 0.0f;

    const float*        urow = u      + (size_t)(m0 + l16) * D_IN;
    const unsigned int* srow = usplit + (size_t)(m0 + l16) * D_IN;

    for (int k0 = 0; k0 < D_IN; k0 += 32) {
        v16bf a_hi, a_lo;
        if (USE_USPLIT) load_a_from_split(srow, k0, half, a_hi, a_lo);
        else            load_a_convert(urow, k0, half, a_hi, a_lo);
#pragma unroll
        for (int t = 0; t < 4; ++t) {
            const unsigned short* wp =
                Bw + (size_t)(n0 + t * 16 + l16) * D_IN + k0 + half * 8;
            v16bf wt = load_w16(wp);
            acc[t] = __builtin_amdgcn_wmma_f32_16x16x32_bf16(
                false, a_lo, false, wt, (short)0, acc[t], false, false);
            acc[t] = __builtin_amdgcn_wmma_f32_16x16x32_bf16(
                false, a_hi, false, wt, (short)0, acc[t], false, false);
        }
    }

    const float sB = *sB_ptr;
#pragma unroll
    for (int t = 0; t < 4; ++t) {
        const int   col = n0 + t * 16 + l16;
        const float ad  = A_diag[col];
#pragma unroll
        for (int r = 0; r < 8; ++r) {
            const int    row = m0 + r + half * 8;
            const size_t idx = (size_t)row * D_STATE + col;
            float xv = __builtin_nontemporal_load(x + idx);  // x read exactly once
            float xn = xv * ad + sB * acc[t][r];
            x_next[idx] = xn;                                // RT: re-read by k2 (L2)
            if (WRITE_XSPLIT) xsplit[idx] = split_pack(xn);  // RT: re-read by k2 (L2)
        }
    }
}

// ---------------------------------------------------------------------------
// Kernel 2: y = s_C*(x_next @ C^T) + s_D*(u @ D^T). Wave -> 16x64 tile.
// ---------------------------------------------------------------------------
template <bool USE_USPLIT, bool USE_XSPLIT>
__global__ void __launch_bounds__(256) ssm_y_kernel(
    const float* __restrict__ xn, const unsigned int* __restrict__ xsplit,
    const float* __restrict__ u,  const unsigned int* __restrict__ usplit,
    const unsigned short* __restrict__ Cw, const unsigned short* __restrict__ Dw,
    const float* __restrict__ sC_ptr, const float* __restrict__ sD_ptr,
    float* __restrict__ y)
{
    const int lane = threadIdx.x & 31;
    const int wave = blockIdx.x * (blockDim.x >> 5) + (threadIdx.x >> 5);
    const int m0   = (wave >> 2) << 4;  // 2048 M-tiles
    const int n0   = (wave & 3) << 6;   // 4 N-groups of 64
    const int half = lane >> 4;
    const int l16  = lane & 15;

    v8f accC[4], accD[4];
#pragma unroll
    for (int t = 0; t < 4; ++t)
#pragma unroll
        for (int r = 0; r < 8; ++r) { accC[t][r] = 0.0f; accD[t][r] = 0.0f; }

    const float*        xrow  = xn     + (size_t)(m0 + l16) * D_STATE;
    const unsigned int* xsrow = xsplit + (size_t)(m0 + l16) * D_STATE;
    for (int k0 = 0; k0 < D_STATE; k0 += 32) {
        v16bf a_hi, a_lo;
        if (USE_XSPLIT) load_a_from_split(xsrow, k0, half, a_hi, a_lo);
        else            load_a_convert(xrow, k0, half, a_hi, a_lo);
#pragma unroll
        for (int t = 0; t < 4; ++t) {
            const unsigned short* wp =
                Cw + (size_t)(n0 + t * 16 + l16) * D_STATE + k0 + half * 8;
            v16bf wt = load_w16(wp);
            accC[t] = __builtin_amdgcn_wmma_f32_16x16x32_bf16(
                false, a_lo, false, wt, (short)0, accC[t], false, false);
            accC[t] = __builtin_amdgcn_wmma_f32_16x16x32_bf16(
                false, a_hi, false, wt, (short)0, accC[t], false, false);
        }
    }

    const float*        urow  = u      + (size_t)(m0 + l16) * D_IN;
    const unsigned int* usrow = usplit + (size_t)(m0 + l16) * D_IN;
    for (int k0 = 0; k0 < D_IN; k0 += 32) {
        v16bf a_hi, a_lo;
        if (USE_USPLIT) load_a_from_split(usrow, k0, half, a_hi, a_lo);
        else            load_a_convert(urow, k0, half, a_hi, a_lo);
#pragma unroll
        for (int t = 0; t < 4; ++t) {
            const unsigned short* wp =
                Dw + (size_t)(n0 + t * 16 + l16) * D_IN + k0 + half * 8;
            v16bf wt = load_w16(wp);
            accD[t] = __builtin_amdgcn_wmma_f32_16x16x32_bf16(
                false, a_lo, false, wt, (short)0, accD[t], false, false);
            accD[t] = __builtin_amdgcn_wmma_f32_16x16x32_bf16(
                false, a_hi, false, wt, (short)0, accD[t], false, false);
        }
    }

    const float sC = *sC_ptr;
    const float sD = *sD_ptr;
#pragma unroll
    for (int t = 0; t < 4; ++t) {
        const int col = n0 + t * 16 + l16;
#pragma unroll
        for (int r = 0; r < 8; ++r) {
            const int row = m0 + r + half * 8;
            // y is never re-read: non-temporal store keeps L2 for x_next stream
            __builtin_nontemporal_store(sC * accC[t][r] + sD * accD[t][r],
                                        y + (size_t)row * D_OUT + col);
        }
    }
}

// ---------------------------------------------------------------------------
extern "C" void kernel_launch(void* const* d_in, const int* in_sizes, int n_in,
                              void* d_out, int out_size, void* d_ws, size_t ws_size,
                              hipStream_t stream) {
    (void)in_sizes; (void)n_in; (void)out_size;

    const float* x     = (const float*)d_in[0];
    const float* u     = (const float*)d_in[1];
    const float* A_bar = (const float*)d_in[2];
    const float* B_bar = (const float*)d_in[3];
    const float* C_bar = (const float*)d_in[4];
    const float* D_bar = (const float*)d_in[5];
    const float* sA    = (const float*)d_in[6];
    const float* sB    = (const float*)d_in[7];
    const float* sC    = (const float*)d_in[8];
    const float* sD    = (const float*)d_in[9];

    char* ws = (char*)d_ws;
    unsigned short* Bw     = (unsigned short*)(ws);                 // 512 KB
    unsigned short* Cw     = (unsigned short*)(ws + 524288);        // 512 KB
    unsigned short* Dw     = (unsigned short*)(ws + 1048576);       // 128 KB
    float*          A_diag = (float*)(ws + 1179648);                // 4 KB

    const size_t OFF_U = (size_t)2 << 20;                       // 2 MB
    const size_t SZ_U  = (size_t)BATCH * D_IN * 4;              // 32 MB
    const size_t OFF_X = OFF_U + SZ_U;
    const size_t SZ_X  = (size_t)BATCH * D_STATE * 4;           // 128 MB
    const bool have_u  = ws_size >= OFF_U + SZ_U;
    const bool have_x  = ws_size >= OFF_X + SZ_X;               // implies room for u
    unsigned int* usplit = (unsigned int*)(ws + OFF_U);
    unsigned int* xsplit = (unsigned int*)(ws + OFF_X);

    float* x_next = (float*)d_out;
    float* yout   = (float*)d_out + (size_t)BATCH * D_STATE;

    ssm_prep_kernel<<<1024, 256, 0, stream>>>(A_bar, B_bar, C_bar, D_bar, sA,
                                              Bw, Cw, Dw, A_diag);
    if (have_u) {
        // 8,388,608 elements, 4 per thread
        ssm_usplit_kernel<<<8192, 256, 0, stream>>>(u, usplit);
    }

    if (have_u && have_x) {
        ssm_xnext_kernel<true, true><<<4096, 256, 0, stream>>>(
            x, u, usplit, Bw, A_diag, sB, x_next, xsplit);
        ssm_y_kernel<true, true><<<1024, 256, 0, stream>>>(
            x_next, xsplit, u, usplit, Cw, Dw, sC, sD, yout);
    } else if (have_u) {
        ssm_xnext_kernel<true, false><<<4096, 256, 0, stream>>>(
            x, u, usplit, Bw, A_diag, sB, x_next, xsplit);
        ssm_y_kernel<true, false><<<1024, 256, 0, stream>>>(
            x_next, xsplit, u, usplit, Cw, Dw, sC, sD, yout);
    } else {
        ssm_xnext_kernel<false, false><<<4096, 256, 0, stream>>>(
            x, u, usplit, Bw, A_diag, sB, x_next, xsplit);
        ssm_y_kernel<false, false><<<1024, 256, 0, stream>>>(
            x_next, xsplit, u, usplit, Cw, Dw, sC, sD, yout);
    }
}
